// GlobalAttn_20023137533955
// MI455X (gfx1250) — compile-verified, matrix operand order checked by reading, not run on hardware
//
#include <hip/hip_runtime.h>
#include <math.h>

typedef __attribute__((ext_vector_type(16))) _Float16 v16h;
typedef __attribute__((ext_vector_type(8)))  _Float16 v8h;
typedef __attribute__((ext_vector_type(4)))  _Float16 v4h;
typedef __attribute__((ext_vector_type(8)))  float    v8f;

#define LDS_STRIDE 264   // halfs per staged edge row: 256 data + 8 pad (528 B -> conflict-free A reads)
#define EDGES_PER_BLOCK 128

// ---------------------------------------------------------------------------
// Kernel 1: Build f16 B-matrix fragments of W1 in the exact CDNA5 wave32 WMMA
// per-lane layout (16-bit B 32x16 tile; element e of lane l holds
// K = e + 8*((e>>3) + (l>>4)), N = l&15).  64 fragments (jt 0..7, kt 0..7),
// each fragment = 32 lanes x 16 halfs (32 B / lane).
// ---------------------------------------------------------------------------
__global__ void ga_prep_bfrag(const float* __restrict__ W1, _Float16* __restrict__ bfrag) {
    int gid = blockIdx.x * blockDim.x + threadIdx.x;   // 0..2047
    if (gid >= 64 * 32) return;
    int lane = gid & 31;
    int fid  = gid >> 5;        // jt*8 + kt
    int jt   = fid >> 3;
    int kt   = fid & 7;
    int n    = lane & 15;
    int hi   = lane >> 4;
    v16h v;
    #pragma unroll
    for (int e = 0; e < 16; ++e) {
        int kk = e + (((e >> 3) + hi) << 3);   // 0..31 within K-tile
        int kr = kt * 32 + kk;                 // row of W1 (0..255)
        int j  = jt * 16 + n;                  // col of W1 (0..127)
        v[e] = (_Float16)W1[kr * 128 + j];
    }
    *(v16h*)(bfrag + (size_t)fid * 512 + lane * 16) = v;   // 32B, 32B-aligned
}

// ---------------------------------------------------------------------------
// Kernel 2: fused [q|k]@W1 + b1 -> leaky_relu -> dot(w2) => attn logits [E,2]
// One block = 128 edges, 8 waves; each wave owns 16 edges.
// ---------------------------------------------------------------------------
__global__ void __launch_bounds__(256)
ga_edge_kernel(const float* __restrict__ q, const float* __restrict__ kmat,
               const float* __restrict__ b1, const float* __restrict__ w2,
               const _Float16* __restrict__ bfrag, float* __restrict__ attn,
               int E) {
    extern __shared__ _Float16 smem[];          // EDGES_PER_BLOCK * LDS_STRIDE halfs
    const int tid = threadIdx.x;
    const long long eBase = (long long)blockIdx.x * EDGES_PER_BLOCK;

    // ---- stage q (cols 0..127) and k (cols 128..255) as f16 into LDS ----
    #pragma unroll
    for (int src = 0; src < 2; ++src) {
        const float* base = src ? kmat : q;
        const int ldsOff = src * 128;
        #pragma unroll
        for (int i = 0; i < 16; ++i) {
            int idx4 = i * 256 + tid;          // 0..4095 float4-units (128 rows * 32)
            int e    = idx4 >> 5;              // edge within block
            int j    = (idx4 & 31) << 2;       // col 0..124 step 4
            if (eBase + e < E) {
                const float4 f = *(const float4*)(base + (eBase + e) * 128 + j);
                v4h h;
                h[0] = (_Float16)f.x; h[1] = (_Float16)f.y;
                h[2] = (_Float16)f.z; h[3] = (_Float16)f.w;
                *(v4h*)(smem + e * LDS_STRIDE + ldsOff + j) = h;
            }
        }
    }
    __syncthreads();

    // ---- per-wave WMMA: 16 edges x 128 outputs, folded straight into attn ----
    const int wv   = tid >> 5;
    const int lane = tid & 31;
    const int m    = lane & 15;      // edge-in-tile selector (A rows / D cols)
    const int hi   = lane >> 4;      // half-wave group
    const int rowA = wv * 16 + m;    // staged row this lane reads for A

    float aH0[8], aH1[8];
    #pragma unroll
    for (int r = 0; r < 8; ++r) { aH0[r] = 0.f; aH1[r] = 0.f; }

    #pragma unroll
    for (int jt = 0; jt < 8; ++jt) {
        const int head = jt >> 2;
        const float b1v = b1[jt * 16 + m];
        const float w2v = w2[head * 64 + (jt & 3) * 16 + m];
        const _Float16* bf = bfrag + (size_t)jt * 8 * 512;

        v8f c = {0.f, 0.f, 0.f, 0.f, 0.f, 0.f, 0.f, 0.f};
        #pragma unroll
        for (int kt = 0; kt < 8; ++kt) {
            // A fragment: two contiguous 8-half chunks per lane from LDS
            const _Float16* arow = smem + rowA * LDS_STRIDE + kt * 32 + hi * 8;
            v8h lo  = *(const v8h*)(arow);
            v8h hi8 = *(const v8h*)(arow + 16);
            v16h A = __builtin_shufflevector(lo, hi8, 0, 1, 2, 3, 4, 5, 6, 7,
                                             8, 9, 10, 11, 12, 13, 14, 15);
            // B fragment: coalesced 32B per lane (L2-resident)
            v16h B = *(const v16h*)(bf + kt * 512 + lane * 16);
            c = __builtin_amdgcn_wmma_f32_16x16x32_f16(
                    false, A, false, B, (short)0, c, false, false);
        }
        // leaky_relu(+b1), scale by w2, reduce the 16 columns per edge
        #pragma unroll
        for (int r = 0; r < 8; ++r) {
            float x = c[r] + b1v;
            x = (x >= 0.f) ? x : 0.01f * x;
            float val = x * w2v;
            val += __shfl_xor(val, 1);
            val += __shfl_xor(val, 2);
            val += __shfl_xor(val, 4);
            val += __shfl_xor(val, 8);   // stays within each 16-lane half
            if (head == 0) aH0[r] += val; else aH1[r] += val;
        }
    }

    if (m == 0) {   // lanes 0 and 16 each write 8 edges
        #pragma unroll
        for (int r = 0; r < 8; ++r) {
            long long e = eBase + wv * 16 + r + hi * 8;
            if (e < E) {
                attn[e * 2 + 0] = aH0[r];
                attn[e * 2 + 1] = aH1[r];
            }
        }
    }
}

// ---------------------------------------------------------------------------
// Kernel 3: per-segment softmax + weighted sum of k.  One block per segment.
// index is sorted -> binary-search the segment range.
// ---------------------------------------------------------------------------
__global__ void __launch_bounds__(128)
ga_segment_kernel(const float* __restrict__ kmat, const int* __restrict__ index,
                  const float* __restrict__ attn, float* __restrict__ out, int E) {
    const int n   = blockIdx.x;
    const int tid = threadIdx.x;
    __shared__ int   sRange[2];
    __shared__ float red0[128], red1[128];
    __shared__ float sStat[4];   // max0, max1, sum0, sum1

    if (tid == 0) {
        int lo = 0, hiE = E;
        while (lo < hiE) { int mid = (lo + hiE) >> 1; if (index[mid] < n) lo = mid + 1; else hiE = mid; }
        sRange[0] = lo;
        int lo2 = lo, hi2 = E;
        while (lo2 < hi2) { int mid = (lo2 + hi2) >> 1; if (index[mid] < n + 1) lo2 = mid + 1; else hi2 = mid; }
        sRange[1] = lo2;
    }
    __syncthreads();
    const int start = sRange[0], end = sRange[1];

    // segment max per head
    float m0 = -3.402823466e38f, m1 = -3.402823466e38f;
    for (int e = start + tid; e < end; e += 128) {
        m0 = fmaxf(m0, attn[2 * e]);
        m1 = fmaxf(m1, attn[2 * e + 1]);
    }
    red0[tid] = m0; red1[tid] = m1; __syncthreads();
    for (int s = 64; s > 0; s >>= 1) {
        if (tid < s) { red0[tid] = fmaxf(red0[tid], red0[tid + s]);
                       red1[tid] = fmaxf(red1[tid], red1[tid + s]); }
        __syncthreads();
    }
    if (tid == 0) { sStat[0] = red0[0]; sStat[1] = red1[0]; }
    __syncthreads();
    m0 = sStat[0]; m1 = sStat[1];

    // segment sum of exp per head
    float s0 = 0.f, s1 = 0.f;
    for (int e = start + tid; e < end; e += 128) {
        s0 += expf(attn[2 * e]     - m0);
        s1 += expf(attn[2 * e + 1] - m1);
    }
    red0[tid] = s0; red1[tid] = s1; __syncthreads();
    for (int s = 64; s > 0; s >>= 1) {
        if (tid < s) { red0[tid] += red0[tid + s]; red1[tid] += red1[tid + s]; }
        __syncthreads();
    }
    if (tid == 0) { sStat[2] = red0[0]; sStat[3] = red1[0]; }
    __syncthreads();

    // weighted sum: thread tid owns output dim tid (head = tid/64)
    const int   h   = tid >> 6;
    const float mh  = sStat[h];
    const float inv = 1.0f / (sStat[2 + h] + 1e-16f);
    float acc = 0.f;
    for (int e = start; e < end; ++e) {
        float w = expf(attn[2 * e + h] - mh) * inv;
        acc += w * kmat[(long long)e * 128 + tid];
    }
    out[(long long)n * 128 + tid] = acc;
}

// ---------------------------------------------------------------------------
extern "C" void kernel_launch(void* const* d_in, const int* in_sizes, int n_in,
                              void* d_out, int out_size, void* d_ws, size_t ws_size,
                              hipStream_t stream) {
    const float* q     = (const float*)d_in[0];
    const float* kmat  = (const float*)d_in[1];
    const int*   index = (const int*)d_in[2];
    // d_in[3] = dim_size (device scalar; N derived host-side from out_size instead)
    const float* W1    = (const float*)d_in[4];
    const float* b1    = (const float*)d_in[5];
    const float* w2    = (const float*)d_in[6];
    float*       out   = (float*)d_out;

    const int E = in_sizes[0] / 128;
    const int N = out_size / 128;

    _Float16* bfrag = (_Float16*)d_ws;                       // 64 KB
    float*    attn  = (float*)((char*)d_ws + 64 * 1024);     // E*2 floats

    ga_prep_bfrag<<<8, 256, 0, stream>>>(W1, bfrag);

    const int nBlocks = (E + EDGES_PER_BLOCK - 1) / EDGES_PER_BLOCK;
    const size_t ldsBytes = (size_t)EDGES_PER_BLOCK * LDS_STRIDE * sizeof(_Float16); // 67584 B
    ga_edge_kernel<<<nBlocks, 256, ldsBytes, stream>>>(q, kmat, b1, w2, bfrag, attn, E);

    ga_segment_kernel<<<N, 128, 0, stream>>>(kmat, index, attn, out, E);
}
